// Attention_45406394253499
// MI455X (gfx1250) — matrix operand order repaired and checked by reading, not compiled, not run on hardware
//
#include <hip/hip_runtime.h>
#include <hip/hip_fp16.h>

// B=8, N=1024, DIM=768, HEADS=12, DHEAD=64, INNER=768
// out = 8192*768 f32, blended = 96*1024*1024 f32 (concatenated in d_out).

typedef __attribute__((ext_vector_type(16))) _Float16 v16h;
typedef __attribute__((ext_vector_type(8)))  float    v8f;
typedef int gvec4i __attribute__((vector_size(16)));
typedef int gvec2i __attribute__((vector_size(8)));

union Frag { v16h h; uint4 q[2]; _Float16 e[16]; };

#define AS1 __attribute__((address_space(1)))
#define AS3 __attribute__((address_space(3)))

#if __has_builtin(__builtin_amdgcn_global_load_async_to_lds_b128)
#define ASYNC_LDS 1
#else
#define ASYNC_LDS 0
#endif

// per-lane 16B global -> LDS copy (async DMA path, ASYNCcnt-tracked)
__device__ __forceinline__ void cp16_g2l(_Float16* dst, const _Float16* src) {
#if ASYNC_LDS
  __builtin_amdgcn_global_load_async_to_lds_b128((AS1 gvec4i*)src,
                                                 (AS3 gvec4i*)dst, 0, 0);
#else
  *(uint4*)dst = *(const uint4*)src;
#endif
}
// per-lane 8B global -> LDS copy
__device__ __forceinline__ void cp8_g2l(_Float16* dst, const _Float16* src) {
#if ASYNC_LDS && __has_builtin(__builtin_amdgcn_global_load_async_to_lds_b64)
  __builtin_amdgcn_global_load_async_to_lds_b64((AS1 gvec2i*)src,
                                                (AS3 gvec2i*)dst, 0, 0);
#else
  *(uint2*)dst = *(const uint2*)src;
#endif
}
__device__ __forceinline__ void async_wait() {
#if ASYNC_LDS
#if __has_builtin(__builtin_amdgcn_s_wait_asynccnt)
  __builtin_amdgcn_s_wait_asynccnt(0);
#else
  asm volatile("s_wait_asynccnt 0" ::: "memory");
#endif
#endif
}

__device__ __forceinline__ v8f wmma_f16(v16h a, v16h b, v8f c) {
  return __builtin_amdgcn_wmma_f32_16x16x32_f16(false, a, false, b, (short)0, c,
                                                false, false);
}

// A-fragment (16x32 f16, MxK). Lane L: M = L&15, K-half offset = (L>>4)*8.
// VGPR j<4: K = koff+2j ; VGPR j>=4: K = 16+koff+2(j-4). Two 16B chunks.
__device__ __forceinline__ v16h load_a_frag(const _Float16* base, int stride,
                                            int mbase, int kbase) {
  int lane = threadIdx.x & 31;
  const _Float16* row = base + (long)(mbase + (lane & 15)) * stride;
  int koff = (lane >> 4) * 8;
  Frag f;
  f.q[0] = *(const uint4*)(row + kbase + koff);
  f.q[1] = *(const uint4*)(row + kbase + 16 + koff);
  return f.h;
}

// B-fragment (32x16 f16, KxN). Lane L: N = L&15, K = (L>>4)*8 + s (s<8), 16 + (L>>4)*8 + (s-8) (s>=8).
// Source indexed as [n][k] with k contiguous (stride in f16 elems per n).
__device__ __forceinline__ v16h load_b_frag(const _Float16* base, long nstride,
                                            int n0, int kbase) {
  int lane = threadIdx.x & 31;
  const _Float16* p = base + (long)(n0 + (lane & 15)) * nstride + kbase +
                      ((lane >> 4) << 3);
  Frag f;
  f.q[0] = *(const uint4*)p;
  f.q[1] = *(const uint4*)(p + 16);
  return f.h;
}

// ---------------- LayerNorm + f16 cast ----------------
__global__ __launch_bounds__(256)
void ln_kernel(const float* __restrict__ x, const float* __restrict__ gamma,
               const float* __restrict__ beta, _Float16* __restrict__ xn) {
  int row = blockIdx.x;
  const float* xr = x + (long)row * 768;
  float s = 0.f, s2 = 0.f;
  for (int i = threadIdx.x; i < 768; i += 256) {
    float v = xr[i]; s += v; s2 += v * v;
  }
  #pragma unroll
  for (int off = 1; off < 32; off <<= 1) {
    s += __shfl_xor(s, off, 32); s2 += __shfl_xor(s2, off, 32);
  }
  __shared__ float ws1[8], ws2[8];
  int w = threadIdx.x >> 5, lane = threadIdx.x & 31;
  if (lane == 0) { ws1[w] = s; ws2[w] = s2; }
  __syncthreads();
  s = 0.f; s2 = 0.f;
  #pragma unroll
  for (int i = 0; i < 8; ++i) { s += ws1[i]; s2 += ws2[i]; }
  float mu = s * (1.f / 768.f);
  float var = s2 * (1.f / 768.f) - mu * mu;
  float rs = rsqrtf(var + 1e-5f);
  for (int i = threadIdx.x; i < 768; i += 256)
    xn[(long)row * 768 + i] = (_Float16)((xr[i] - mu) * rs * gamma[i] + beta[i]);
}

__global__ __launch_bounds__(256)
void cvt_kernel(const float* __restrict__ s, _Float16* __restrict__ d, int n) {
  int stride = gridDim.x * 256;
  for (int i = blockIdx.x * 256 + threadIdx.x; i < n; i += stride)
    d[i] = (_Float16)s[i];
}

// ---------------- Tiled f16 GEMM (M x 768) @ (768 x ldb) ----------------
// Block tile 128x64, 8 waves in 4(M)x2(N); each wave 32x32 = 2x2 WMMA tiles.
// mode 0: QKV -> scatter q[bh][n][d], k[bh][n][d], v^T[bh][d][n] (f16)
// mode 1: out = A@W + bias (f32)
__global__ __launch_bounds__(256)
void gemm_kernel(const _Float16* __restrict__ A, const _Float16* __restrict__ Bw,
                 int ldb, int mode,
                 _Float16* __restrict__ qf, _Float16* __restrict__ kf,
                 _Float16* __restrict__ vt,
                 const float* __restrict__ bias, float* __restrict__ outp) {
  __shared__ __align__(16) _Float16 As[128 * 40];
  __shared__ __align__(16) _Float16 BsT[64 * 40];
  int tid = threadIdx.x;
  int w = tid >> 5, lane = tid & 31;
  int wm = w & 3, wn = w >> 2;
  int n = lane & 15, half = lane >> 4;
  int blockRow = blockIdx.y * 128;
  int blockCol = blockIdx.x * 64;
  v8f acc[2][2] = {};

  for (int kb = 0; kb < 768; kb += 32) {
    {  // A tile 128x32 -> LDS row-major (stride 40), async DMA if available
      int r = tid >> 1;
      int c = (tid & 1) << 4;
      const _Float16* src = A + (long)(blockRow + r) * 768 + kb + c;
      cp16_g2l(&As[r * 40 + c], src);
      cp16_g2l(&As[r * 40 + c + 8], src + 8);
    }
    {  // B tile 32x64 -> LDS transposed [n][k] (stride 40)
      int kk = tid >> 3;
      int n0 = (tid & 7) << 3;
      Frag f;
      f.q[0] = *(const uint4*)&Bw[(long)(kb + kk) * ldb + blockCol + n0];
      #pragma unroll
      for (int i = 0; i < 8; ++i) BsT[(n0 + i) * 40 + kk] = f.e[i];
    }
    async_wait();
    __syncthreads();
    v16h a0 = load_a_frag(As, 40, wm * 32, 0);
    v16h a1 = load_a_frag(As, 40, wm * 32 + 16, 0);
    v16h b0 = load_b_frag(BsT, 40, wn * 32, 0);
    v16h b1 = load_b_frag(BsT, 40, wn * 32 + 16, 0);
    acc[0][0] = wmma_f16(a0, b0, acc[0][0]);
    acc[0][1] = wmma_f16(a0, b1, acc[0][1]);
    acc[1][0] = wmma_f16(a1, b0, acc[1][0]);
    acc[1][1] = wmma_f16(a1, b1, acc[1][1]);
    __syncthreads();
  }

  #pragma unroll
  for (int tm = 0; tm < 2; ++tm)
    #pragma unroll
    for (int tn = 0; tn < 2; ++tn) {
      v8f a = acc[tm][tn];
      #pragma unroll
      for (int r = 0; r < 8; ++r) {
        int row = blockRow + wm * 32 + tm * 16 + r + 8 * half;
        int col = blockCol + wn * 32 + tn * 16 + n;
        float val = a[r];
        if (mode == 0) {
          int which = col >= 1536 ? 2 : (col >= 768 ? 1 : 0);
          int ic = col - which * 768;
          int head = ic >> 6, d = ic & 63;
          int bb = row >> 10, nn = row & 1023;
          long bh = (long)(bb * 12 + head);
          if (which == 0)      qf[(bh * 1024 + nn) * 64 + d] = (_Float16)val;
          else if (which == 1) kf[(bh * 1024 + nn) * 64 + d] = (_Float16)val;
          else                 vt[(bh * 64 + d) * 1024 + nn] = (_Float16)val;
        } else {
          __builtin_nontemporal_store(val + bias[col],
                                      &outp[(long)row * 768 + col]);
        }
      }
    }
}

// ---------------- Attention: q@kT, blend+store, softmax, attn@v ----------------
// Block = (16 rows) x (one b,h). 8 waves; wave w owns columns [w*128, w*128+128).
__global__ __launch_bounds__(256)
void attn_kernel(const _Float16* __restrict__ qf, const _Float16* __restrict__ kf,
                 const _Float16* __restrict__ vt, const float* __restrict__ h,
                 float* __restrict__ blended, _Float16* __restrict__ aout) {
  constexpr float SCALE = 0.125f;  // 64^-0.5
  int row0 = blockIdx.x * 16;
  int bh = blockIdx.y;
  int bq = bh / 12, head = bh % 12;
  int tid = threadIdx.x;
  int w = tid >> 5, lane = tid & 31;
  int n = lane & 15, half = lane >> 4;
  int colbase = w * 128;

  __shared__ __align__(16) _Float16 qtile[16 * 72];
  __shared__ __align__(16) _Float16 pbuf[8][16 * 136];
  __shared__ float outacc[16 * 64];
  __shared__ float red[8][16];
  __shared__ float rowv[16];

  {  // stage Q tile 16x64 into LDS (async DMA if available)
    int r = tid >> 4;
    int c = (tid & 15) << 2;
    cp8_g2l(&qtile[r * 72 + c], qf + ((long)bh * 1024 + row0 + r) * 64 + c);
  }
  for (int i = tid; i < 1024; i += 256) outacc[i] = 0.f;
  async_wait();
  __syncthreads();

  v16h qa0 = load_a_frag(qtile, 72, 0, 0);
  v16h qa1 = load_a_frag(qtile, 72, 0, 32);

  const _Float16* kp = kf + (long)bh * 1024 * 64;
  const float* hp = h + ((long)bh * 1024 + row0) * 1024;
  float blv[8][8];

  #pragma unroll
  for (int j = 0; j < 8; ++j) {  // q @ kT + blend (16 x 128 per wave)
    int n0 = colbase + j * 16;
    v8f acc = {};
    acc = wmma_f16(qa0, load_b_frag(kp, 64, n0, 0), acc);
    acc = wmma_f16(qa1, load_b_frag(kp, 64, n0, 32), acc);
    #pragma unroll
    for (int r = 0; r < 8; ++r) {
      int row = r + 8 * half;
      // h streamed once: non-temporal load; blended streamed once: NT store
      float hv = __builtin_nontemporal_load(&hp[row * 1024 + n0 + n]);
      float v = 0.25f * (acc[r] * SCALE) + 0.75f * hv;
      blv[j][r] = v;
      __builtin_nontemporal_store(
          v, &blended[((long)bh * 1024 + row0 + row) * 1024 + n0 + n]);
    }
  }

  // ---- row max across 1024 cols ----
  float pr[8];
  #pragma unroll
  for (int r = 0; r < 8; ++r) {
    float m = blv[0][r];
    #pragma unroll
    for (int j = 1; j < 8; ++j) m = fmaxf(m, blv[j][r]);
    #pragma unroll
    for (int off = 1; off < 16; off <<= 1) m = fmaxf(m, __shfl_xor(m, off, 32));
    pr[r] = m;
  }
  if (lane == 0)       for (int r = 0; r < 8; ++r) red[w][r] = pr[r];
  else if (lane == 16) for (int r = 0; r < 8; ++r) red[w][8 + r] = pr[r];
  __syncthreads();
  if (tid < 16) {
    float m = red[0][tid];
    for (int ww = 1; ww < 8; ++ww) m = fmaxf(m, red[ww][tid]);
    rowv[tid] = m;
  }
  __syncthreads();
  float rmax[8];
  #pragma unroll
  for (int r = 0; r < 8; ++r) rmax[r] = rowv[r + 8 * half];

  // ---- exp + row sum ----
  #pragma unroll
  for (int r = 0; r < 8; ++r) {
    float s = 0.f;
    #pragma unroll
    for (int j = 0; j < 8; ++j) {
      float e = __expf(blv[j][r] - rmax[r]);
      blv[j][r] = e; s += e;
    }
    #pragma unroll
    for (int off = 1; off < 16; off <<= 1) s += __shfl_xor(s, off, 32);
    pr[r] = s;
  }
  if (lane == 0)       for (int r = 0; r < 8; ++r) red[w][r] = pr[r];
  else if (lane == 16) for (int r = 0; r < 8; ++r) red[w][8 + r] = pr[r];
  __syncthreads();
  if (tid < 16) {
    float s = red[0][tid];
    for (int ww = 1; ww < 8; ++ww) s += red[ww][tid];
    rowv[tid] = s;
  }
  __syncthreads();
  float rinv[8];
  #pragma unroll
  for (int r = 0; r < 8; ++r) rinv[r] = 1.0f / rowv[r + 8 * half];

  // ---- store attn f16 to LDS in A-fragment layout ----
  _Float16* pw = pbuf[w];
  #pragma unroll
  for (int j = 0; j < 8; ++j)
    #pragma unroll
    for (int r = 0; r < 8; ++r)
      pw[(r + 8 * half) * 136 + j * 16 + n] = (_Float16)(blv[j][r] * rinv[r]);
  __syncthreads();

  // ---- attn @ V (per-wave partial over 128 cols) ----
  const _Float16* vp = vt + (long)bh * 64 * 1024;
  v8f oacc[4] = {};
  #pragma unroll
  for (int kt = 0; kt < 4; ++kt) {
    v16h af = load_a_frag(pw, 136, 0, kt * 32);
    #pragma unroll
    for (int dt = 0; dt < 4; ++dt)
      oacc[dt] = wmma_f16(af, load_b_frag(vp, 1024, dt * 16, colbase + kt * 32),
                          oacc[dt]);
  }
  #pragma unroll
  for (int dt = 0; dt < 4; ++dt)
    #pragma unroll
    for (int r = 0; r < 8; ++r)
      atomicAdd(&outacc[(r + 8 * half) * 64 + dt * 16 + n], oacc[dt][r]);
  __syncthreads();

  for (int i = tid; i < 1024; i += 256) {
    int row = i >> 6, d = i & 63;
    aout[((long)bq * 1024 + row0 + row) * 768 + head * 64 + d] =
        (_Float16)outacc[i];
  }
}

extern "C" void kernel_launch(void* const* d_in, const int* in_sizes, int n_in,
                              void* d_out, int out_size, void* d_ws, size_t ws_size,
                              hipStream_t stream) {
  const float* x     = (const float*)d_in[0];
  const float* h     = (const float*)d_in[1];
  const float* gamma = (const float*)d_in[2];
  const float* beta  = (const float*)d_in[3];
  const float* w_qkv = (const float*)d_in[4];
  const float* w_out = (const float*)d_in[5];
  const float* b_out = (const float*)d_in[6];
  float* outp    = (float*)d_out;            // 8192*768
  float* blended = outp + (long)8192 * 768;  // 96*1024*1024

  // workspace carve-up (f16), ~67.6 MB total
  _Float16* xnf   = (_Float16*)d_ws;
  _Float16* wqkvf = xnf + (long)8192 * 768;
  _Float16* woutf = wqkvf + (long)768 * 2304;
  _Float16* qff   = woutf + (long)768 * 768;
  _Float16* kff   = qff + (long)96 * 1024 * 64;
  _Float16* vtf   = kff + (long)96 * 1024 * 64;
  _Float16* aoutf = vtf + (long)96 * 1024 * 64;

  cvt_kernel<<<1728, 256, 0, stream>>>(w_qkv, wqkvf, 768 * 2304);
  cvt_kernel<<<576, 256, 0, stream>>>(w_out, woutf, 768 * 768);
  ln_kernel<<<8192, 256, 0, stream>>>(x, gamma, beta, xnf);
  gemm_kernel<<<dim3(36, 64), 256, 0, stream>>>(xnf, wqkvf, 2304, 0,
                                                qff, kff, vtf, nullptr, nullptr);
  attn_kernel<<<dim3(64, 96), 256, 0, stream>>>(qff, kff, vtf, h, blended, aoutf);
  gemm_kernel<<<dim3(12, 64), 256, 0, stream>>>(aoutf, woutf, 768, 1,
                                                nullptr, nullptr, nullptr,
                                                b_out, outp);
}
